// MOELayer_8083128451228
// MI455X (gfx1250) — matrix operand order repaired and checked by reading, not compiled, not run on hardware
//
#include <hip/hip_runtime.h>
#include <hip/hip_bf16.h>
#include <math.h>

// MoE layer for MI455X (gfx1250): bf16 WMMA expert GEMMs with f32 accumulate.
// Needs ~105 MB of d_ws scratch.

#define T_TOK  4096
#define DMODEL 1024
#define HDIM   4096
#define NEXP   8
#define CAP    1280

#if defined(__gfx1250__) && __has_builtin(__builtin_amdgcn_global_load_async_to_lds_b128) && \
    __has_builtin(__builtin_amdgcn_s_wait_asynccnt)
#define MOE_ASYNC_LDS 1
#else
#define MOE_ASYNC_LDS 0
#endif

typedef float  f32x8 __attribute__((ext_vector_type(8)));
typedef int    v4i   __attribute__((ext_vector_type(4)));
typedef __bf16 bfx4  __attribute__((ext_vector_type(4)));
typedef __bf16 bfx8  __attribute__((ext_vector_type(8)));
typedef __bf16 bfx16 __attribute__((ext_vector_type(16)));

typedef __attribute__((address_space(1))) v4i* gv4i_ptr;   // global int4*
typedef __attribute__((address_space(3))) v4i* lv4i_ptr;   // LDS int4*

union BF16x16 { bfx16 v; bfx8 h[2]; };

// ---------------------------------------------------------------- init
__global__ __launch_bounds__(256) void moe_init(float* __restrict__ out,
                                                int* __restrict__ slot_token) {
    size_t i = (size_t)blockIdx.x * 256 + threadIdx.x;
    if (i < (size_t)T_TOK * DMODEL / 4) {
        float4 z; z.x = z.y = z.z = z.w = 0.f;
        ((float4*)out)[i] = z;
    }
    if (i < (size_t)NEXP * CAP) slot_token[i] = -1;
}

// ---------------------------------------------------------------- router
// one wave per token; w_g (1024x8) cached in LDS
__global__ __launch_bounds__(256) void moe_router(const float* __restrict__ x,
                                                  const float* __restrict__ wg,
                                                  int* __restrict__ topk_idx,
                                                  float* __restrict__ topk_w) {
    __shared__ float swg[DMODEL * NEXP];
    for (int i = threadIdx.x; i < DMODEL * NEXP; i += 256) swg[i] = wg[i];
    __syncthreads();
    const int wave = threadIdx.x >> 5;
    const int lane = threadIdx.x & 31;
    const int t = blockIdx.x * 8 + wave;
    const float* xr = x + (size_t)t * DMODEL;
    float acc[NEXP];
#pragma unroll
    for (int e = 0; e < NEXP; ++e) acc[e] = 0.f;
    for (int j = lane; j < DMODEL; j += 32) {
        float xv = xr[j];
#pragma unroll
        for (int e = 0; e < NEXP; ++e) acc[e] += xv * swg[j * NEXP + e];
    }
#pragma unroll
    for (int off = 16; off >= 1; off >>= 1)
#pragma unroll
        for (int e = 0; e < NEXP; ++e) acc[e] += __shfl_xor(acc[e], off, 32);
    if (lane == 0) {
        int i1 = 0; float v1 = acc[0];
        for (int e = 1; e < NEXP; ++e) if (acc[e] > v1) { v1 = acc[e]; i1 = e; }
        int i2 = (i1 == 0) ? 1 : 0; float v2 = acc[i2];
        for (int e = 0; e < NEXP; ++e)
            if (e != i1 && acc[e] > v2) { v2 = acc[e]; i2 = e; }
        float ez = expf(v2 - v1);       // v2 <= v1
        float inv = 1.f / (1.f + ez);
        topk_idx[t * 2 + 0] = i1;
        topk_idx[t * 2 + 1] = i2;
        topk_w[t * 2 + 0] = inv;
        topk_w[t * 2 + 1] = ez * inv;
    }
}

// ---------------------------------------------------------------- rank / capacity drop
// Exclusive prefix counts over the k-major flattened (k*T) ordering, as in the
// reference cumsum. Single block, 256 threads x 32 entries each.
__global__ __launch_bounds__(256) void moe_rank(const int* __restrict__ topk_idx,
                                                const float* __restrict__ topk_w,
                                                int* __restrict__ slot_token,
                                                float* __restrict__ slot_weight) {
    __shared__ int scan[256][NEXP];
    const int tid = threadIdx.x;
    int cnt[NEXP];
#pragma unroll
    for (int e = 0; e < NEXP; ++e) cnt[e] = 0;
    const int base = tid * 32;
    for (int i = 0; i < 32; ++i) {
        int f = base + i;
        int t = f & (T_TOK - 1);
        int k = f >> 12;                 // T_TOK == 4096
        cnt[topk_idx[t * 2 + k]]++;
    }
    int run[NEXP];
#pragma unroll
    for (int e = 0; e < NEXP; ++e) { run[e] = cnt[e]; scan[tid][e] = cnt[e]; }
    for (int off = 1; off < 256; off <<= 1) {
        __syncthreads();
        int prev[NEXP]; bool has = tid >= off;
        if (has) {
#pragma unroll
            for (int e = 0; e < NEXP; ++e) prev[e] = scan[tid - off][e];
        }
        __syncthreads();
        if (has) {
#pragma unroll
            for (int e = 0; e < NEXP; ++e) { run[e] += prev[e]; scan[tid][e] = run[e]; }
        }
    }
    int ex[NEXP];
#pragma unroll
    for (int e = 0; e < NEXP; ++e) ex[e] = run[e] - cnt[e];   // exclusive prefix
    for (int i = 0; i < 32; ++i) {
        int f = base + i;
        int t = f & (T_TOK - 1);
        int k = f >> 12;
        int e = topk_idx[t * 2 + k];
        int r = ex[e]++;
        if (r < CAP) {
            int s = e * CAP + r;
            slot_token[s]  = t;
            slot_weight[s] = topk_w[t * 2 + k];
        }
    }
}

// ---------------------------------------------------------------- dispatch gather (f32 -> bf16)
__global__ __launch_bounds__(128) void moe_gather(const float* __restrict__ x,
                                                  const int* __restrict__ slot_token,
                                                  __bf16* __restrict__ xb) {
    const int slot = blockIdx.x;
    const int t = slot_token[slot];
    __bf16* dst = xb + (size_t)slot * DMODEL;
    const int j0 = threadIdx.x * 8;
    bfx8 v;
    if (t >= 0) {
        const float* src = x + (size_t)t * DMODEL + j0;
        float4 a = *(const float4*)(src);
        float4 b = *(const float4*)(src + 4);
        v[0] = (__bf16)a.x; v[1] = (__bf16)a.y; v[2] = (__bf16)a.z; v[3] = (__bf16)a.w;
        v[4] = (__bf16)b.x; v[5] = (__bf16)b.y; v[6] = (__bf16)b.z; v[7] = (__bf16)b.w;
    } else {
#pragma unroll
        for (int i = 0; i < 8; ++i) v[i] = (__bf16)0.f;
    }
    *(bfx8*)(dst + j0) = v;
}

// ---------------------------------------------------------------- WMMA GEMM core
// C(128x256) per block, 256 threads = 8 waves in a 2x4 grid, wave tile 64x64,
// 16 accumulators (v8f) per wave. A: bf16 [M x lda]. B: f32 [lda x ldbn],
// converted to bf16 while staging (transposed) into LDS.
// A-tile staging uses GLOBAL_LOAD_ASYNC_TO_LDS_B128 (ASYNCcnt) when available.
__device__ __forceinline__ void gemm_tile_mma(const __bf16* __restrict__ Ab,
                                              const float* __restrict__ Bb,
                                              int lda, int ldbn, int tm, int tn,
                                              __bf16 (*lA)[40], __bf16 (*lB)[40],
                                              f32x8 acc[4][4]) {
    const int tid  = threadIdx.x;
    const int wave = tid >> 5;
    const int lane = tid & 31;
    const int wm = (wave >> 2) * 64;
    const int wn = (wave & 3) * 64;
    const int g  = lane >> 4;
    const int ln = lane & 15;

    for (int kk = 0; kk < lda; kk += 32) {
        __syncthreads();
        // A tile: 128x32 bf16, row-major, b128 chunks (async to LDS when possible)
#pragma unroll
        for (int l = 0; l < 2; ++l) {
            int idx = tid + l * 256;
            int row = idx >> 2;
            int ch  = idx & 3;
            const __bf16* gsrc = Ab + (size_t)(tm + row) * lda + kk + ch * 8;
            __bf16* ldst = &lA[row][ch * 8];
#if MOE_ASYNC_LDS
            __builtin_amdgcn_global_load_async_to_lds_b128(
                (gv4i_ptr)gsrc, (lv4i_ptr)ldst, 0, 0);
#else
            *(bfx8*)ldst = *(const bfx8*)gsrc;
#endif
        }
        // B tile: 32x256 f32 -> bf16, stored transposed lB[n][k]
#pragma unroll
        for (int l = 0; l < 8; ++l) {
            int task = tid + l * 256;
            int n  = task & 255;
            int kc = task >> 8;
            const float* src = Bb + (size_t)(kk + kc * 4) * ldbn + tn + n;
            bfx4 bv;
            bv.x = (__bf16)src[0];
            bv.y = (__bf16)src[(size_t)ldbn];
            bv.z = (__bf16)src[(size_t)2 * ldbn];
            bv.w = (__bf16)src[(size_t)3 * ldbn];
            *(bfx4*)&lB[n][kc * 4] = bv;
        }
        if (kk + 32 < lda) {   // hint next K tile (global_prefetch_b8)
            __builtin_prefetch(Ab + (size_t)(tm + (tid >> 1)) * lda + kk + 32, 0, 1);
            __builtin_prefetch(Bb + (size_t)(kk + 32 + (tid & 31)) * ldbn + tn +
                               ((tid >> 5) << 5), 0, 1);
        }
#if MOE_ASYNC_LDS
        __builtin_amdgcn_s_wait_asynccnt(0);
#endif
        __syncthreads();

        // fragment loads (layouts per ISA 7.12.2): two b128 per fragment
        BF16x16 afr[4], bfr[4];
#pragma unroll
        for (int mi = 0; mi < 4; ++mi) {
            int rm = wm + mi * 16 + ln;
            afr[mi].h[0] = *(const bfx8*)&lA[rm][8 * g];        // K = 8g..8g+7
            afr[mi].h[1] = *(const bfx8*)&lA[rm][16 + 8 * g];   // K = 16+8g..23+8g
        }
#pragma unroll
        for (int ni = 0; ni < 4; ++ni) {
            int cn = wn + ni * 16 + ln;
            bfr[ni].h[0] = *(const bfx8*)&lB[cn][8 * g];        // K = 8g..8g+7
            bfr[ni].h[1] = *(const bfx8*)&lB[cn][16 + 8 * g];   // K = 16+8g..23+8g
        }
#pragma unroll
        for (int mi = 0; mi < 4; ++mi)
#pragma unroll
            for (int ni = 0; ni < 4; ++ni)
                acc[mi][ni] = __builtin_amdgcn_wmma_f32_16x16x32_bf16(
                    false, afr[mi].v, false, bfr[ni].v,
                    (short)0, acc[mi][ni], false, false);
    }
}

// ---------------------------------------------------------------- GEMM1 + exact GELU
__global__ __launch_bounds__(256) void moe_gemm1(const __bf16* __restrict__ xb,
                                                 const float* __restrict__ cfc,
                                                 __bf16* __restrict__ hbuf) {
    __shared__ __bf16 lA[128][40];
    __shared__ __bf16 lB[256][40];
    const int e  = blockIdx.z;
    const int tm = blockIdx.y * 128;
    const int tn = blockIdx.x * 256;
    f32x8 acc[4][4];
#pragma unroll
    for (int mi = 0; mi < 4; ++mi)
#pragma unroll
        for (int ni = 0; ni < 4; ++ni)
#pragma unroll
            for (int r = 0; r < 8; ++r) acc[mi][ni][r] = 0.f;

    gemm_tile_mma(xb + (size_t)e * CAP * DMODEL,
                  cfc + (size_t)e * DMODEL * HDIM,
                  DMODEL, HDIM, tm, tn, lA, lB, acc);

    const int wave = threadIdx.x >> 5;
    const int lane = threadIdx.x & 31;
    const int wm = (wave >> 2) * 64;
    const int wn = (wave & 3) * 64;
    const int g  = lane >> 4;
    const int ln = lane & 15;
    __bf16* hb = hbuf + (size_t)e * CAP * HDIM;
#pragma unroll
    for (int mi = 0; mi < 4; ++mi)
#pragma unroll
        for (int ni = 0; ni < 4; ++ni)
#pragma unroll
            for (int r = 0; r < 8; ++r) {
                int row = tm + wm + mi * 16 + 8 * g + r;
                int col = tn + wn + ni * 16 + ln;
                float v  = acc[mi][ni][r];
                float ge = 0.5f * v * (1.0f + erff(v * 0.70710678118654752f));
                hb[(size_t)row * HDIM + col] = (__bf16)ge;
            }
}

// ---------------------------------------------------------------- GEMM2 + fused weighted combine
__global__ __launch_bounds__(256) void moe_gemm2(const __bf16* __restrict__ hbuf,
                                                 const float* __restrict__ cproj,
                                                 const int* __restrict__ slot_token,
                                                 const float* __restrict__ slot_weight,
                                                 float* __restrict__ out) {
    __shared__ __bf16 lA[128][40];
    __shared__ __bf16 lB[256][40];
    const int e  = blockIdx.z;
    const int tm = blockIdx.y * 128;
    const int tn = blockIdx.x * 256;
    f32x8 acc[4][4];
#pragma unroll
    for (int mi = 0; mi < 4; ++mi)
#pragma unroll
        for (int ni = 0; ni < 4; ++ni)
#pragma unroll
            for (int r = 0; r < 8; ++r) acc[mi][ni][r] = 0.f;

    gemm_tile_mma(hbuf + (size_t)e * CAP * HDIM,
                  cproj + (size_t)e * HDIM * DMODEL,
                  HDIM, DMODEL, tm, tn, lA, lB, acc);

    const int wave = threadIdx.x >> 5;
    const int lane = threadIdx.x & 31;
    const int wm = (wave >> 2) * 64;
    const int wn = (wave & 3) * 64;
    const int g  = lane >> 4;
    const int ln = lane & 15;
#pragma unroll
    for (int mi = 0; mi < 4; ++mi)
#pragma unroll
        for (int ni = 0; ni < 4; ++ni)
#pragma unroll
            for (int r = 0; r < 8; ++r) {
                int row  = tm + wm + mi * 16 + 8 * g + r;
                int col  = tn + wn + ni * 16 + ln;
                int slot = e * CAP + row;
                int t    = slot_token[slot];
                if (t >= 0) {
                    float w = slot_weight[slot];
                    atomicAdd(out + (size_t)t * DMODEL + col, w * acc[mi][ni][r]);
                }
            }
}

// ---------------------------------------------------------------- launch
extern "C" void kernel_launch(void* const* d_in, const int* in_sizes, int n_in,
                              void* d_out, int out_size, void* d_ws, size_t ws_size,
                              hipStream_t stream) {
    (void)in_sizes; (void)n_in; (void)out_size; (void)ws_size;
    const float* x     = (const float*)d_in[0];   // [T, 1024]
    const float* wg    = (const float*)d_in[1];   // [1024, 8]
    const float* cfc   = (const float*)d_in[2];   // [8, 1024, 4096]
    const float* cproj = (const float*)d_in[3];   // [8, 4096, 1024]
    float* out = (float*)d_out;                   // [T, 1024]

    char* ws = (char*)d_ws;
    int*    topk_idx    = (int*)  (ws + 0);                       // 32 KB
    float*  topk_w      = (float*)(ws + 32768);                   // 32 KB
    int*    slot_token  = (int*)  (ws + 65536);                   // 40 KB
    float*  slot_weight = (float*)(ws + 106496);                  // 40 KB
    __bf16* xb          = (__bf16*)(ws + 147456);                 // 20 MB
    __bf16* hbuf        = (__bf16*)(ws + 147456 +
                                    (size_t)NEXP * CAP * DMODEL * 2); // 80 MB

    moe_init  <<<4096, 256, 0, stream>>>(out, slot_token);
    moe_router<<<T_TOK / 8, 256, 0, stream>>>(x, wg, topk_idx, topk_w);
    moe_rank  <<<1, 256, 0, stream>>>(topk_idx, topk_w, slot_token, slot_weight);
    moe_gather<<<NEXP * CAP, 128, 0, stream>>>(x, slot_token, xb);
    moe_gemm1 <<<dim3(HDIM / 256, CAP / 128, NEXP), 256, 0, stream>>>(xb, cfc, hbuf);
    moe_gemm2 <<<dim3(DMODEL / 256, CAP / 128, NEXP), 256, 0, stream>>>(hbuf, cproj,
                                                                        slot_token,
                                                                        slot_weight, out);
}
